// LSTMTagger_CRF_56590489092782
// MI455X (gfx1250) — matrix operand order repaired and checked by reading, not compiled, not run on hardware
//
#include <hip/hip_runtime.h>
#include <hip/hip_bf16.h>
#include <math.h>

typedef _Float16 v16h __attribute__((ext_vector_type(16)));
typedef _Float16 v8h  __attribute__((ext_vector_type(8)));
typedef _Float16 h2v  __attribute__((ext_vector_type(2)));
typedef float    v8f  __attribute__((ext_vector_type(8)));
typedef float    v4f  __attribute__((ext_vector_type(4)));
typedef unsigned int v4u __attribute__((ext_vector_type(4)));
typedef int      v8i  __attribute__((ext_vector_type(8)));
typedef int      v4i  __attribute__((ext_vector_type(4)));

#define LSEQ 4096
#define EDIM 300
#define KP0  320      // E padded to multiple of 32
#define HDIM 512
#define GDIM 2048     // 4*H
#define K1   1024     // 2*H
#define NT   24
#define NTP  32       // T padded to multiple of 16

#if defined(__gfx1250__) && __has_builtin(__builtin_amdgcn_tensor_load_to_lds)
#define HAVE_TDM 1
#endif

// ---------------- weight / activation conversion kernels ----------------

// fp32 [Nv][K] row-major -> fp16 [Ntot][Kp] row-major, zero padded
__global__ void k_cvt_pad(const float* __restrict__ src, _Float16* __restrict__ dst,
                          int Nv, int K, int Kp, int Ntot) {
  int idx = blockIdx.x * blockDim.x + threadIdx.x;
  if (idx >= Ntot * Kp) return;
  int n = idx / Kp, k = idx - n * Kp;
  float v = (n < Nv && k < K) ? src[(size_t)n * K + k] : 0.f;
  dst[idx] = (_Float16)v;
}

// w_hh fp32 [G][H] -> fp16 transposed [H][G] (coalesced recurrent reads)
__global__ void k_cvt_T(const float* __restrict__ src, _Float16* __restrict__ dst) {
  int idx = blockIdx.x * blockDim.x + threadIdx.x;
  if (idx >= HDIM * GDIM) return;
  int k = idx / GDIM, n = idx - k * GDIM;
  dst[idx] = (_Float16)src[(size_t)n * HDIM + k];
}

// embedding gather into K-major fp16 transpose: x0t[k][m]
__global__ void k_embed(const float* __restrict__ emb, const int* __restrict__ sent,
                        _Float16* __restrict__ x0t) {
  int idx = blockIdx.x * blockDim.x + threadIdx.x;
  if (idx >= KP0 * LSEQ) return;
  int k = idx / LSEQ, m = idx - k * LSEQ;
  float v = (k < EDIM) ? emb[(size_t)sent[m] * EDIM + k] : 0.f;
  x0t[idx] = (_Float16)v;
}

// concat [hf | hb] into K-major fp16 transpose: xt[k][m], k in [0,1024)
__global__ void k_build_xt(const float* __restrict__ hsF, const float* __restrict__ hsB,
                           _Float16* __restrict__ xt) {
  int idx = blockIdx.x * blockDim.x + threadIdx.x;
  if (idx >= K1 * LSEQ) return;
  int k = idx / LSEQ, m = idx - k * LSEQ;
  float v = (k < HDIM) ? hsF[(size_t)m * HDIM + k]
                       : hsB[(size_t)m * HDIM + (k - HDIM)];
  xt[idx] = (_Float16)v;
}

// ---------------- WMMA GEMM: Out[m][n] = sum_k W[n][k]*Xt[k][m] + bias[n] ----
__global__ void __launch_bounds__(128)
k_gemm_wmma(const _Float16* __restrict__ W, int Kp,
            const _Float16* __restrict__ Xt,
            const float* __restrict__ bias,
            float* __restrict__ Out, int ldo,
            int Ntot, int Nstore) {
  int lane = threadIdx.x & 31;
  int wv = blockIdx.x * (blockDim.x >> 5) + (threadIdx.x >> 5);
  int ntiles = Ntot >> 4;
  int tn = wv % ntiles;
  int tm = wv / ntiles;
  int n0 = tn << 4, m0 = tm << 4;
  int mr = lane & 15, hi = lane >> 4;

  v8f c;
#pragma unroll
  for (int r = 0; r < 8; ++r) c[r] = 0.f;

  const _Float16* ap = W + (size_t)(n0 + mr) * Kp + hi * 8;
  const _Float16* bp = Xt + (size_t)(hi * 8) * LSEQ + m0 + mr;

  for (int k0 = 0; k0 < Kp; k0 += 32) {
    // A fragment: lane<16 -> K {0..7,16..23}, lane>=16 -> K {8..15,24..31}
    v8h alo = *(const v8h*)(ap + k0);
    v8h ahi = *(const v8h*)(ap + k0 + 16);
    v16h a = __builtin_shufflevector(alo, ahi, 0,1,2,3,4,5,6,7,8,9,10,11,12,13,14,15);
    // B fragment: column = lane&15, lane<16 -> K {0..7,16..23}, lane>=16 -> K {8..15,24..31}
    v16h b;
#pragma unroll
    for (int s = 0; s < 8; ++s) {
      b[s]     = bp[(size_t)(k0 + s) * LSEQ];
      b[8 + s] = bp[(size_t)(k0 + 16 + s) * LSEQ];
    }
    c = __builtin_amdgcn_wmma_f32_16x16x32_f16(false, a, false, b,
                                               (short)0, c, false, false);
  }

  int m = m0 + mr;  // D: N = lane&15
#pragma unroll
  for (int r = 0; r < 8; ++r) {
    int n = n0 + hi * 8 + r;  // D: M = r + 8*hi
    if (n < Nstore)
      Out[(size_t)m * ldo + n] = c[r] + bias[n];
  }
}

// ---------------- TDM helper: DMA one fp32 row of GDIM elems into LDS -------
#ifdef HAVE_TDM
__device__ __forceinline__ void tdm_load_row(const float* gsrc, float* ldsdst) {
  unsigned lds = (unsigned)(size_t)ldsdst;              // low 32 bits = LDS offset
  unsigned long long ga = (unsigned long long)(size_t)gsrc;
  // D# group0: count=1 (valid user descriptor), lds_addr, global_addr, type=2
  v4u g0 = { 1u, lds, (unsigned)ga,
             (unsigned)((ga >> 32) & 0x1FFFFFFu) | (2u << 30) };
  // D# group1: data_size=4B(code 2); tensor_dim0=GDIM; tensor_dim1=1;
  // tile_dim0=GDIM (1-D tile); tile_dim1/2 unused; dim0_stride=GDIM
  v8i g1 = { (int)(2u << 16),
             (int)(((unsigned)GDIM) << 16),
             (int)(1u << 16),
             (int)(((unsigned)GDIM) << 16),
             0,
             (int)GDIM,
             0, 0 };
  v4i z4 = { 0, 0, 0, 0 };
#if __has_include(<hip/amd_detail/amd_gfx1250_TDM.h>)
  v8i z8 = { 0, 0, 0, 0, 0, 0, 0, 0 };
  __builtin_amdgcn_tensor_load_to_lds(g0, g1, z4, z4, z8, 0);   // 6-arg toolchain
#else
  __builtin_amdgcn_tensor_load_to_lds(g0, g1, z4, z4, 0);       // 5-arg toolchain
#endif
}
#endif

// ---------------- persistent recurrent scan (one workgroup per direction) ----
// thread t owns gate outputs 2t and 2t+1 (adjacent -> single 4B weight load)
__global__ void __launch_bounds__(1024)
k_lstm(const float* __restrict__ preF, const float* __restrict__ preB,
       const _Float16* __restrict__ wtF, const _Float16* __restrict__ wtB,
       float* __restrict__ hsF, float* __restrict__ hsB) {
  const bool rev = (blockIdx.x == 1);
  const float*    pre = rev ? preB : preF;
  const _Float16* wt  = rev ? wtB : wtF;   // [H][G] fp16 (transposed W_hh)
  float*          hs  = rev ? hsB : hsF;

  __shared__ __align__(16) float hsh[HDIM];
  __shared__ __align__(16) float zsh[GDIM];
#ifdef HAVE_TDM
  __shared__ __align__(16) float prelds[2][GDIM];
#endif

  int t = threadIdx.x;                       // 0..1023
  int wid = __builtin_amdgcn_readfirstlane((int)(threadIdx.x >> 5)); // uniform wave id
  const _Float16* wcol = wt + 2 * t;         // column pair base
  float cc = 0.f;
  if (t < HDIM) hsh[t] = 0.f;

#ifdef HAVE_TDM
  if (wid == 0) {                            // wave 0 stages first pre row
    int tt0 = rev ? (LSEQ - 1) : 0;
    tdm_load_row(pre + (size_t)tt0 * GDIM, &prelds[0][0]);
    __builtin_amdgcn_s_wait_tensorcnt(0);
  }
#endif
  __syncthreads();

  for (int s = 0; s < LSEQ; ++s) {
    int tt = rev ? (LSEQ - 1 - s) : s;
#ifdef HAVE_TDM
    int buf = s & 1;
    if (wid == 0 && (s + 1) < LSEQ) {        // overlap next row DMA with compute
      int tn2 = rev ? (LSEQ - 2 - s) : (s + 1);
      tdm_load_row(pre + (size_t)tn2 * GDIM, &prelds[1 - buf][0]);
    }
    float a0 = prelds[buf][2 * t];
    float a1 = prelds[buf][2 * t + 1];
#else
    const float* pr = pre + (size_t)tt * GDIM;
    float a0 = pr[2 * t], a1 = pr[2 * t + 1];
    if (s + 1 < LSEQ) {
      int tn2 = rev ? (LSEQ - 2 - s) : (s + 1);
      __builtin_prefetch(pre + (size_t)tn2 * GDIM + 2 * t, 0, 1);
    }
#endif

#pragma unroll 2
    for (int k = 0; k < HDIM; k += 4) {
      v4f h4 = *(const v4f*)&hsh[k];
      h2v w0 = *(const h2v*)(wcol + (size_t)(k + 0) * GDIM);
      h2v w1 = *(const h2v*)(wcol + (size_t)(k + 1) * GDIM);
      h2v w2 = *(const h2v*)(wcol + (size_t)(k + 2) * GDIM);
      h2v w3 = *(const h2v*)(wcol + (size_t)(k + 3) * GDIM);
      a0 = fmaf((float)w0.x, h4.x, a0);  a1 = fmaf((float)w0.y, h4.x, a1);
      a0 = fmaf((float)w1.x, h4.y, a0);  a1 = fmaf((float)w1.y, h4.y, a1);
      a0 = fmaf((float)w2.x, h4.z, a0);  a1 = fmaf((float)w2.y, h4.z, a1);
      a0 = fmaf((float)w3.x, h4.w, a0);  a1 = fmaf((float)w3.y, h4.w, a1);
    }
    zsh[2 * t]     = a0;
    zsh[2 * t + 1] = a1;
    __syncthreads();

    if (t < HDIM) {
      float zi = zsh[t], zf = zsh[HDIM + t], zg = zsh[2 * HDIM + t], zo = zsh[3 * HDIM + t];
      float ig = 1.f / (1.f + __expf(-zi));
      float fg = 1.f / (1.f + __expf(-zf));
      float og = 1.f / (1.f + __expf(-zo));
      cc = fg * cc + ig * tanhf(zg);
      float hn = og * tanhf(cc);
      hsh[t] = hn;
      hs[(size_t)tt * HDIM + t] = hn;
    }
#ifdef HAVE_TDM
    if (wid == 0) __builtin_amdgcn_s_wait_tensorcnt(0);  // next buffer resident
#endif
    __syncthreads();
  }
}

// ---------------- CRF negative log-likelihood ----------------
__global__ void __launch_bounds__(64)
k_crf(const float* __restrict__ em, const int* __restrict__ tags,
      const float* __restrict__ start, const float* __restrict__ endt,
      const float* __restrict__ trans, float* __restrict__ out) {
  __shared__ float al[NT];
  __shared__ float tr[NT * NT];
  __shared__ float numsh;
  int j = threadIdx.x;
  for (int i = j; i < NT * NT; i += blockDim.x) tr[i] = trans[i];
  __syncthreads();

  if (j == 0) {
    float num = start[tags[0]] + endt[tags[LSEQ - 1]];
    for (int t = 0; t < LSEQ; ++t) num += em[(size_t)t * NT + tags[t]];
    for (int t = 1; t < LSEQ; ++t) num += tr[tags[t - 1] * NT + tags[t]];
    numsh = num;
  }
  if (j < NT) al[j] = start[j] + em[j];
  __syncthreads();

  for (int t = 1; t < LSEQ; ++t) {
    float na = 0.f;
    if (j < NT) {
      float mx = -1e30f;
#pragma unroll
      for (int i = 0; i < NT; ++i) mx = fmaxf(mx, al[i] + tr[i * NT + j]);
      float sm = 0.f;
#pragma unroll
      for (int i = 0; i < NT; ++i) sm += __expf(al[i] + tr[i * NT + j] - mx);
      na = mx + __logf(sm) + em[(size_t)t * NT + j];
    }
    __syncthreads();
    if (j < NT) al[j] = na;
    __syncthreads();
  }

  if (j == 0) {
    float mx = -1e30f;
    for (int i = 0; i < NT; ++i) mx = fmaxf(mx, al[i] + endt[i]);
    float sm = 0.f;
    for (int i = 0; i < NT; ++i) sm += __expf(al[i] + endt[i] - mx);
    out[0] = (mx + __logf(sm)) - numsh;
  }
}

// ---------------- launcher ----------------
extern "C" void kernel_launch(void* const* d_in, const int* in_sizes, int n_in,
                              void* d_out, int out_size, void* d_ws, size_t ws_size,
                              hipStream_t stream) {
  const float* emb       = (const float*)d_in[0];
  const float* w_ih_l0f  = (const float*)d_in[1];
  const float* w_hh_l0f  = (const float*)d_in[2];
  const float* b_l0f     = (const float*)d_in[3];
  const float* w_ih_l0b  = (const float*)d_in[4];
  const float* w_hh_l0b  = (const float*)d_in[5];
  const float* b_l0b     = (const float*)d_in[6];
  const float* w_ih_l1f  = (const float*)d_in[7];
  const float* w_hh_l1f  = (const float*)d_in[8];
  const float* b_l1f     = (const float*)d_in[9];
  const float* w_ih_l1b  = (const float*)d_in[10];
  const float* w_hh_l1b  = (const float*)d_in[11];
  const float* b_l1b     = (const float*)d_in[12];
  const float* w_out     = (const float*)d_in[13];
  const float* b_out     = (const float*)d_in[14];
  const float* start_tr  = (const float*)d_in[15];
  const float* end_tr    = (const float*)d_in[16];
  const float* trans     = (const float*)d_in[17];
  const int*   sentence  = (const int*)d_in[18];
  const int*   tags      = (const int*)d_in[19];
  float* out = (float*)d_out;

  char* ws = (char*)d_ws;
  size_t off = 0;
  auto carve = [&](size_t bytes) -> char* {
    off = (off + 255) & ~(size_t)255;
    char* p = ws + off;
    off += bytes;
    return p;
  };

  _Float16* x0t   = (_Float16*)carve((size_t)KP0 * LSEQ * 2);
  _Float16* xt    = (_Float16*)carve((size_t)K1 * LSEQ * 2);   // x1t, reused as x2t
  _Float16* whl0f = (_Float16*)carve((size_t)GDIM * KP0 * 2);
  _Float16* whl0b = (_Float16*)carve((size_t)GDIM * KP0 * 2);
  _Float16* whl1f = (_Float16*)carve((size_t)GDIM * K1 * 2);
  _Float16* whl1b = (_Float16*)carve((size_t)GDIM * K1 * 2);
  _Float16* whout = (_Float16*)carve((size_t)NTP * K1 * 2);
  _Float16* wt0f  = (_Float16*)carve((size_t)HDIM * GDIM * 2);
  _Float16* wt0b  = (_Float16*)carve((size_t)HDIM * GDIM * 2);
  _Float16* wt1f  = (_Float16*)carve((size_t)HDIM * GDIM * 2);
  _Float16* wt1b  = (_Float16*)carve((size_t)HDIM * GDIM * 2);
  float*    pre_f = (float*)carve((size_t)LSEQ * GDIM * 4);    // reused layer0/1
  float*    pre_b = (float*)carve((size_t)LSEQ * GDIM * 4);
  float*    hs_f  = (float*)carve((size_t)LSEQ * HDIM * 4);    // reused layer0/1
  float*    hs_b  = (float*)carve((size_t)LSEQ * HDIM * 4);
  float*    emis  = (float*)carve((size_t)LSEQ * NT * 4);

  const int thr = 256;
  auto blk = [&](long n) { return (int)((n + thr - 1) / thr); };

  // weight conversion
  k_cvt_pad<<<blk((long)GDIM * KP0), thr, 0, stream>>>(w_ih_l0f, whl0f, GDIM, EDIM, KP0, GDIM);
  k_cvt_pad<<<blk((long)GDIM * KP0), thr, 0, stream>>>(w_ih_l0b, whl0b, GDIM, EDIM, KP0, GDIM);
  k_cvt_pad<<<blk((long)GDIM * K1),  thr, 0, stream>>>(w_ih_l1f, whl1f, GDIM, K1, K1, GDIM);
  k_cvt_pad<<<blk((long)GDIM * K1),  thr, 0, stream>>>(w_ih_l1b, whl1b, GDIM, K1, K1, GDIM);
  k_cvt_pad<<<blk((long)NTP * K1),   thr, 0, stream>>>(w_out,    whout, NT, K1, K1, NTP);
  k_cvt_T<<<blk((long)HDIM * GDIM), thr, 0, stream>>>(w_hh_l0f, wt0f);
  k_cvt_T<<<blk((long)HDIM * GDIM), thr, 0, stream>>>(w_hh_l0b, wt0b);
  k_cvt_T<<<blk((long)HDIM * GDIM), thr, 0, stream>>>(w_hh_l1f, wt1f);
  k_cvt_T<<<blk((long)HDIM * GDIM), thr, 0, stream>>>(w_hh_l1b, wt1b);

  // embedding gather (transposed fp16)
  k_embed<<<blk((long)KP0 * LSEQ), thr, 0, stream>>>(emb, sentence, x0t);

  const int blocksG = (GDIM / 16) * (LSEQ / 16) / 4;  // 8192, one wave per tile
  const int blocksE = (NTP / 16) * (LSEQ / 16) / 4;   // 128

  // layer 0
  k_gemm_wmma<<<blocksG, 128, 0, stream>>>(whl0f, KP0, x0t, b_l0f, pre_f, GDIM, GDIM, GDIM);
  k_gemm_wmma<<<blocksG, 128, 0, stream>>>(whl0b, KP0, x0t, b_l0b, pre_b, GDIM, GDIM, GDIM);
  k_lstm<<<2, 1024, 0, stream>>>(pre_f, pre_b, wt0f, wt0b, hs_f, hs_b);
  k_build_xt<<<blk((long)K1 * LSEQ), thr, 0, stream>>>(hs_f, hs_b, xt);

  // layer 1
  k_gemm_wmma<<<blocksG, 128, 0, stream>>>(whl1f, K1, xt, b_l1f, pre_f, GDIM, GDIM, GDIM);
  k_gemm_wmma<<<blocksG, 128, 0, stream>>>(whl1b, K1, xt, b_l1b, pre_b, GDIM, GDIM, GDIM);
  k_lstm<<<2, 1024, 0, stream>>>(pre_f, pre_b, wt1f, wt1b, hs_f, hs_b);
  k_build_xt<<<blk((long)K1 * LSEQ), thr, 0, stream>>>(hs_f, hs_b, xt);

  // emissions + CRF
  k_gemm_wmma<<<blocksE, 128, 0, stream>>>(whout, K1, xt, b_out, emis, NT, NTP, NT);
  k_crf<<<1, 64, 0, stream>>>(emis, tags, start_tr, end_tr, trans, out);
}
